// Mamba2Attention_8211977470178
// MI455X (gfx1250) — matrix operand order repaired and checked by reading, not compiled, not run on hardware
//
#include <hip/hip_runtime.h>

// ---------------------------------------------------------------------------
// Mamba2Attention block for MI455X (gfx1250, wave32, WMMA 16x16x32 f16).
// f32 -> f16 convert pass, then an all-f16 data path with f32 accumulation:
//   cvt(x, Wq..Wo) -> gemm_tn (Q,K,V, f16 out) -> head_mlp (q,k in place)
//   -> flash attention (f16 out) -> gemm_tn (Wo, f32 out)
// GEMM staging uses GLOBAL_LOAD_ASYNC_TO_LDS_B128 (ASYNCcnt-tracked) with a
// double-buffered BK=64 main loop; synchronous fallback if builtin missing.
// ---------------------------------------------------------------------------

typedef __attribute__((ext_vector_type(16))) _Float16 v16h;
typedef __attribute__((ext_vector_type(8)))  _Float16 v8h;
typedef __attribute__((ext_vector_type(8)))  float    v8f;
typedef __attribute__((ext_vector_type(4)))  float    v4f;
typedef __attribute__((ext_vector_type(4)))  int      v4i;

#define D_MODEL  1024
#define N_HEADS  16
#define HEAD_DIM 64
#define BATCH    2
#define SEQ      2048
#define NTOK     (BATCH * SEQ)        // 4096
#define SCALE    0.125f               // 64^-0.5
#define FNEG     (-3.0e38f)

// ---- fast reciprocal ------------------------------------------------------
#if __has_builtin(__builtin_amdgcn_rcpf)
#define FAST_RCP(x) __builtin_amdgcn_rcpf(x)
#else
#define FAST_RCP(x) (1.0f / (x))
#endif

// ---- async copy to LDS (CDNA5) -------------------------------------------
#if __has_builtin(__builtin_amdgcn_global_load_async_to_lds_b128)
#define USE_ASYNC 1
#else
#define USE_ASYNC 0
#endif

#if USE_ASYNC
#if __has_builtin(__builtin_amdgcn_s_wait_asynccnt)
#define ASYNC_WAIT(n) __builtin_amdgcn_s_wait_asynccnt(n)
#else
#define ASYNC_WAIT(n) asm volatile("s_wait_asynccnt %0" ::"n"(n) : "memory")
#endif
#else
#define ASYNC_WAIT(n) ((void)0)
#endif

typedef __attribute__((address_space(1))) v4i gas_v4i;  // global
typedef __attribute__((address_space(3))) v4i las_v4i;  // LDS

__device__ __forceinline__ void cp16(_Float16* lds, const _Float16* g) {
#if USE_ASYNC
  __builtin_amdgcn_global_load_async_to_lds_b128(
      (gas_v4i*)(unsigned long long)(__SIZE_TYPE__)g,
      (las_v4i*)(unsigned)(__SIZE_TYPE__)lds, 0, 0);
#else
  *(v8h*)lds = *(const v8h*)g;
#endif
}

// ---- WMMA helper ----------------------------------------------------------
__device__ __forceinline__ v8f wmma16(v16h a, v16h b, v8f c) {
  return __builtin_amdgcn_wmma_f32_16x16x32_f16(false, a, false, b, (short)0, c,
                                                false, false);
}

// A operand: 16x32 f16 tile, row-major in LDS (row stride ld halves).
__device__ __forceinline__ v16h frag_a(const _Float16* base, int ld) {
  const int lane = threadIdx.x & 31;
  const _Float16* p = base + (lane & 15) * ld + (lane >> 4) * 8;
  v8h lo = *(const v8h*)(p);
  v8h hi = *(const v8h*)(p + 16);
  v16h r;
#pragma unroll
  for (int i = 0; i < 8; ++i) { r[i] = lo[i]; r[i + 8] = hi[i]; }
  return r;
}

// B operand: 32x16 (KxN); source tile stored transposed [N][K] in LDS.
__device__ __forceinline__ v16h frag_b(const _Float16* base, int ld) {
  const int lane = threadIdx.x & 31;
  const _Float16* p = base + (lane & 15) * ld + (lane >> 4) * 8;
  v8h lo = *(const v8h*)(p);
  v8h hi = *(const v8h*)(p + 16);
  v16h r;
#pragma unroll
  for (int i = 0; i < 8; ++i) { r[i] = lo[i]; r[i + 8] = hi[i]; }
  return r;
}

__device__ __forceinline__ void zero8(v8f& v) {
#pragma unroll
  for (int i = 0; i < 8; ++i) v[i] = 0.0f;
}

// ---------------------------------------------------------------------------
// f32 -> f16 conversion pass
// ---------------------------------------------------------------------------
__global__ __launch_bounds__(256) void cvt_f16(const float* __restrict__ s,
                                               _Float16* __restrict__ d,
                                               int n) {
  int i = (blockIdx.x * 256 + threadIdx.x) * 8;
  if (i + 8 <= n) {
    v4f a = *(const v4f*)(s + i);
    v4f b = *(const v4f*)(s + i + 4);
    v8h o;
#pragma unroll
    for (int j = 0; j < 4; ++j) {
      o[j] = (_Float16)a[j];
      o[j + 4] = (_Float16)b[j];
    }
    *(v8h*)(d + i) = o;
  }
}

// ---------------------------------------------------------------------------
// GEMM: C[M,N] = A[M,K] * Bt[N,K]^T, f16 in, OutT out. Compile-time dims.
// Block tile 128x128, 8 waves each 32x64, BK=64 (2 WMMA k-steps per stage),
// double-buffered LDS fed by async-to-LDS copies.
// ---------------------------------------------------------------------------
#define GBM 128
#define GBN 128
#define GBK 64
#define GLD 72  // padded LDS row stride (halves)

template <int M, int N, int K, typename OutT>
__global__ __launch_bounds__(256) void gemm_tn(const _Float16* __restrict__ A,
                                               const _Float16* __restrict__ Bt,
                                               OutT* __restrict__ C) {
  __shared__ __align__(16) _Float16 sA[2][GBM * GLD];
  __shared__ __align__(16) _Float16 sB[2][GBN * GLD];

  const int t    = threadIdx.x;
  const int wave = t >> 5;
  const int lane = t & 31;
  const int hi   = lane >> 4;
  const int col  = lane & 15;
  const int m0   = blockIdx.y * GBM;
  const int n0   = blockIdx.x * GBN;
  const int wm   = (wave & 3) * 32;
  const int wn   = (wave >> 2) * 64;

  auto stage = [&](int k0, int buf) {
#pragma unroll
    for (int c = 0; c < 4; ++c) {
      int chunk = t + c * 256;  // 1024 chunks of 8 halves per tile
      int row   = chunk >> 3;
      int kc    = (chunk & 7) * 8;
      cp16(&sA[buf][row * GLD + kc], A + (size_t)(m0 + row) * K + k0 + kc);
      cp16(&sB[buf][row * GLD + kc], Bt + (size_t)(n0 + row) * K + k0 + kc);
    }
  };

  v8f acc[2][4];
#pragma unroll
  for (int a = 0; a < 2; ++a)
#pragma unroll
    for (int b = 0; b < 4; ++b) zero8(acc[a][b]);

  stage(0, 0);
  constexpr int NKT = K / GBK;
  for (int kt = 0; kt < NKT; ++kt) {
    const int cur = kt & 1;
    if (kt + 1 < NKT) {
      stage((kt + 1) * GBK, cur ^ 1);
      ASYNC_WAIT(8);  // 8 async ops outstanding for the next tile
    } else {
      ASYNC_WAIT(0);
    }
    __syncthreads();

#pragma unroll
    for (int ks = 0; ks < 2; ++ks) {
      v16h af[2], bf[4];
#pragma unroll
      for (int mi = 0; mi < 2; ++mi)
        af[mi] = frag_a(&sA[cur][(wm + mi * 16) * GLD + ks * 32], GLD);
#pragma unroll
      for (int ni = 0; ni < 4; ++ni)
        bf[ni] = frag_b(&sB[cur][(wn + ni * 16) * GLD + ks * 32], GLD);
#pragma unroll
      for (int mi = 0; mi < 2; ++mi)
#pragma unroll
        for (int ni = 0; ni < 4; ++ni)
          acc[mi][ni] = wmma16(af[mi], bf[ni], acc[mi][ni]);
    }

    __syncthreads();
  }

  // epilogue: constant N -> row strides fold into store immediates
#pragma unroll
  for (int mi = 0; mi < 2; ++mi)
#pragma unroll
    for (int ni = 0; ni < 4; ++ni) {
      OutT* cp = C + (size_t)(m0 + wm + mi * 16 + hi * 8) * N +
                 (n0 + wn + ni * 16 + col);
#pragma unroll
      for (int i = 0; i < 8; ++i) cp[i * N] = (OutT)acc[mi][ni][i];
    }
}

// ---------------------------------------------------------------------------
// Per-head SSM MLP, f16 in place: H += silu(H*W1^T + b1) * W2^T + b2
// 8 waves * 16 rows = 128 head-rows per block; 2 chained WMMA GEMMs (K=64).
// ---------------------------------------------------------------------------
__global__ __launch_bounds__(256) void head_mlp(
    _Float16* __restrict__ H, const float* __restrict__ W1,
    const float* __restrict__ b1, const float* __restrict__ W2,
    const float* __restrict__ b2) {
  __shared__ __align__(16) _Float16 sW1[64 * 72];
  __shared__ __align__(16) _Float16 sW2[64 * 72];
  __shared__ __align__(16) _Float16 sH[8][16 * 72];
  __shared__ float sb1[64], sb2[64];

  const int t    = threadIdx.x;
  const int wave = t >> 5;
  const int lane = t & 31;
  const int hi   = lane >> 4;
  const int col  = lane & 15;

  for (int i = t; i < 64 * 64; i += 256) {
    int r = i >> 6, c = i & 63;
    sW1[r * 72 + c] = (_Float16)W1[i];
    sW2[r * 72 + c] = (_Float16)W2[i];
  }
  if (t < 64) { sb1[t] = b1[t]; sb2[t] = b2[t]; }

  const size_t r0 = (size_t)blockIdx.x * 128 + wave * 16;
  for (int i = lane; i < 16 * 8; i += 32) {  // 128 chunks of 8 halves
    int rr = i >> 3, cc = (i & 7) * 8;
    *(v8h*)&sH[wave][rr * 72 + cc] = *(const v8h*)(H + (r0 + rr) * 64 + cc);
  }
  __syncthreads();

  v8f acc[4];
#pragma unroll
  for (int ni = 0; ni < 4; ++ni) zero8(acc[ni]);
#pragma unroll
  for (int ks = 0; ks < 2; ++ks) {
    v16h a = frag_a(&sH[wave][ks * 32], 72);
#pragma unroll
    for (int ni = 0; ni < 4; ++ni)
      acc[ni] = wmma16(a, frag_b(sW1 + ni * 16 * 72 + ks * 32, 72), acc[ni]);
  }

#pragma unroll
  for (int ni = 0; ni < 4; ++ni) {
    int c = ni * 16 + col;
    float bb = sb1[c];
#pragma unroll
    for (int i = 0; i < 8; ++i) {
      float x = acc[ni][i] + bb;
      float s = x * FAST_RCP(1.0f + __expf(-x));  // SiLU via v_rcp_f32
      sH[wave][(i + hi * 8) * 72 + c] = (_Float16)s;
    }
  }

  v8f acc2[4];
#pragma unroll
  for (int ni = 0; ni < 4; ++ni) zero8(acc2[ni]);
#pragma unroll
  for (int ks = 0; ks < 2; ++ks) {
    v16h a = frag_a(&sH[wave][ks * 32], 72);
#pragma unroll
    for (int ni = 0; ni < 4; ++ni)
      acc2[ni] = wmma16(a, frag_b(sW2 + ni * 16 * 72 + ks * 32, 72), acc2[ni]);
  }

#pragma unroll
  for (int ni = 0; ni < 4; ++ni) {
    int c = ni * 16 + col;
    float bb = sb2[c];
#pragma unroll
    for (int i = 0; i < 8; ++i) {
      _Float16* p = H + (r0 + i + hi * 8) * 64 + c;
      *p = (_Float16)((float)*p + acc2[ni][i] + bb);
    }
  }
}

// ---------------------------------------------------------------------------
// Flash attention, causal, f16 q/k/v/out, f32 online softmax.
// Grid (S/128, N_HEADS, B), 8 waves x 16 q-rows, 64-key tiles.
// ---------------------------------------------------------------------------
#define QT 128
#define KTL 64

__global__ __launch_bounds__(256) void flash_attn(
    const _Float16* __restrict__ Q, const _Float16* __restrict__ Kb,
    const _Float16* __restrict__ V, _Float16* __restrict__ O) {
  __shared__ __align__(16) _Float16 sQ[QT * 72];
  __shared__ __align__(16) _Float16 sK[KTL * 72];
  __shared__ __align__(16) _Float16 sVT[HEAD_DIM * 72];
  __shared__ __align__(16) _Float16 sP[8][16 * 72];

  const int t    = threadIdx.x;
  const int wave = t >> 5;
  const int lane = t & 31;
  const int hi   = lane >> 4;
  const int col  = lane & 15;
  const int qt   = blockIdx.x;
  const int h    = blockIdx.y;
  const int b    = blockIdx.z;
  const size_t base = (size_t)b * SEQ * D_MODEL + (size_t)h * HEAD_DIM;

  // stage Q block: 128 x 64 halves (async when available)
  for (int i = t; i < QT * 8; i += 256) {
    int r = i >> 3, c = (i & 7) * 8;
    cp16(sQ + r * 72 + c, Q + base + (size_t)(qt * QT + r) * D_MODEL + c);
  }

  float mrow[8], lrow[8];
  v8f accO[4];
#pragma unroll
  for (int i = 0; i < 8; ++i) { mrow[i] = FNEG; lrow[i] = 0.0f; }
#pragma unroll
  for (int ni = 0; ni < 4; ++ni) zero8(accO[ni]);

  const int qrow0  = qt * QT + wave * 16;
  const int ktiles = (qt * QT + QT) / KTL;

  for (int kt = 0; kt < ktiles; ++kt) {
    __syncthreads();
    for (int i = t; i < KTL * 8; i += 256) {
      int r = i >> 3, c = (i & 7) * 8;
      cp16(sK + r * 72 + c, Kb + base + (size_t)(kt * KTL + r) * D_MODEL + c);
    }
    for (int i = t; i < KTL * HEAD_DIM; i += 256) {  // transpose V -> [d][key]
      int key = i >> 6, d = i & 63;
      sVT[d * 72 + key] = V[base + (size_t)(kt * KTL + key) * D_MODEL + d];
    }
    ASYNC_WAIT(0);
    __syncthreads();

    if (kt * KTL <= qrow0 + 15) {
      v8f s[4];
#pragma unroll
      for (int ni = 0; ni < 4; ++ni) zero8(s[ni]);
#pragma unroll
      for (int ks = 0; ks < 2; ++ks) {
        v16h a = frag_a(sQ + (wave * 16) * 72 + ks * 32, 72);
#pragma unroll
        for (int ni = 0; ni < 4; ++ni)
          s[ni] = wmma16(a, frag_b(sK + ni * 16 * 72 + ks * 32, 72), s[ni]);
      }

      float tm[8];
#pragma unroll
      for (int i = 0; i < 8; ++i) tm[i] = FNEG;
#pragma unroll
      for (int ni = 0; ni < 4; ++ni) {
        int kg = kt * KTL + ni * 16 + col;
#pragma unroll
        for (int i = 0; i < 8; ++i) {
          int qg = qrow0 + i + hi * 8;
          float x = s[ni][i] * SCALE;
          if (kg > qg) x = FNEG;
          s[ni][i] = x;
          tm[i] = fmaxf(tm[i], x);
        }
      }
#pragma unroll
      for (int i = 0; i < 8; ++i) {
        tm[i] = fmaxf(tm[i], __shfl_xor(tm[i], 1, 32));
        tm[i] = fmaxf(tm[i], __shfl_xor(tm[i], 2, 32));
        tm[i] = fmaxf(tm[i], __shfl_xor(tm[i], 4, 32));
        tm[i] = fmaxf(tm[i], __shfl_xor(tm[i], 8, 32));
      }

      float sf[8], rs[8];
#pragma unroll
      for (int i = 0; i < 8; ++i) {
        float mn = fmaxf(mrow[i], tm[i]);
        sf[i]   = __expf(mrow[i] - mn);
        mrow[i] = mn;
        rs[i]   = 0.0f;
      }

#pragma unroll
      for (int ni = 0; ni < 4; ++ni) {
#pragma unroll
        for (int i = 0; i < 8; ++i) {
          float p = __expf(s[ni][i] - mrow[i]);
          rs[i] += p;
          sP[wave][(i + hi * 8) * 72 + ni * 16 + col] = (_Float16)p;
        }
      }
#pragma unroll
      for (int i = 0; i < 8; ++i) {
        rs[i] += __shfl_xor(rs[i], 1, 32);
        rs[i] += __shfl_xor(rs[i], 2, 32);
        rs[i] += __shfl_xor(rs[i], 4, 32);
        rs[i] += __shfl_xor(rs[i], 8, 32);
        lrow[i] = lrow[i] * sf[i] + rs[i];
      }
#pragma unroll
      for (int ni = 0; ni < 4; ++ni)
#pragma unroll
        for (int i = 0; i < 8; ++i) accO[ni][i] *= sf[i];

#pragma unroll
      for (int ks = 0; ks < 2; ++ks) {
        v16h p = frag_a(&sP[wave][ks * 32], 72);
#pragma unroll
        for (int ni = 0; ni < 4; ++ni)
          accO[ni] =
              wmma16(p, frag_b(sVT + ni * 16 * 72 + ks * 32, 72), accO[ni]);
      }
    }
  }

  // normalize + store [b, q, h, d] (reciprocal hoisted per row)
  float inv[8];
#pragma unroll
  for (int i = 0; i < 8; ++i)
    inv[i] = lrow[i] > 0.0f ? FAST_RCP(lrow[i]) : 0.0f;
#pragma unroll
  for (int ni = 0; ni < 4; ++ni) {
    int d = ni * 16 + col;
#pragma unroll
    for (int i = 0; i < 8; ++i) {
      int qg = qrow0 + i + hi * 8;
      O[base + (size_t)qg * D_MODEL + d] = (_Float16)(accO[ni][i] * inv[i]);
    }
  }
}

// ---------------------------------------------------------------------------
// Host-side launcher
// ---------------------------------------------------------------------------
extern "C" void kernel_launch(void* const* d_in, const int* in_sizes, int n_in,
                              void* d_out, int out_size, void* d_ws,
                              size_t ws_size, hipStream_t stream) {
  (void)in_sizes; (void)n_in; (void)out_size; (void)ws_size;
  const float* x   = (const float*)d_in[0];
  // d_in[1] = attn_mask (causal, handled analytically)
  const float* Wq  = (const float*)d_in[2];
  const float* Wk  = (const float*)d_in[3];
  const float* Wv  = (const float*)d_in[4];
  const float* Wo  = (const float*)d_in[5];
  const float* qW1 = (const float*)d_in[6];
  const float* qb1 = (const float*)d_in[7];
  const float* qW2 = (const float*)d_in[8];
  const float* qb2 = (const float*)d_in[9];
  const float* kW1 = (const float*)d_in[10];
  const float* kb1 = (const float*)d_in[11];
  const float* kW2 = (const float*)d_in[12];
  const float* kb2 = (const float*)d_in[13];

  const size_t mat  = (size_t)NTOK * D_MODEL;     // 4M elements
  const size_t wmat = (size_t)D_MODEL * D_MODEL;  // 1M elements
  _Float16* xh  = (_Float16*)d_ws;
  _Float16* wqh = xh + mat;
  _Float16* wkh = wqh + wmat;
  _Float16* wvh = wkh + wmat;
  _Float16* woh = wvh + wmat;
  _Float16* q   = woh + wmat;
  _Float16* k   = q + mat;
  _Float16* v   = k + mat;
  _Float16* ao  = v + mat;

  dim3 blk(256);
  cvt_f16<<<dim3((int)(mat / 2048)), blk, 0, stream>>>(x, xh, (int)mat);
  cvt_f16<<<dim3((int)(wmat / 2048)), blk, 0, stream>>>(Wq, wqh, (int)wmat);
  cvt_f16<<<dim3((int)(wmat / 2048)), blk, 0, stream>>>(Wk, wkh, (int)wmat);
  cvt_f16<<<dim3((int)(wmat / 2048)), blk, 0, stream>>>(Wv, wvh, (int)wmat);
  cvt_f16<<<dim3((int)(wmat / 2048)), blk, 0, stream>>>(Wo, woh, (int)wmat);

  dim3 gproj(D_MODEL / GBN, NTOK / GBM);  // (8, 32)
  gemm_tn<NTOK, D_MODEL, D_MODEL, _Float16>
      <<<gproj, blk, 0, stream>>>(xh, wqh, q);
  gemm_tn<NTOK, D_MODEL, D_MODEL, _Float16>
      <<<gproj, blk, 0, stream>>>(xh, wkh, k);
  gemm_tn<NTOK, D_MODEL, D_MODEL, _Float16>
      <<<gproj, blk, 0, stream>>>(xh, wvh, v);

  dim3 gmlp((NTOK * N_HEADS) / 128);  // 512 blocks of 128 head-rows
  head_mlp<<<gmlp, blk, 0, stream>>>(q, qW1, qb1, qW2, qb2);
  head_mlp<<<gmlp, blk, 0, stream>>>(k, kW1, kb1, kW2, kb2);

  dim3 gattn(SEQ / QT, N_HEADS, BATCH);  // (16, 16, 2)
  flash_attn<<<gattn, blk, 0, stream>>>(q, k, v, ao);

  gemm_tn<NTOK, D_MODEL, D_MODEL, float>
      <<<gproj, blk, 0, stream>>>(ao, woh, (float*)d_out);
}
